// DiffusionLM_38989713113106
// MI455X (gfx1250) — matrix operand order repaired and checked by reading, not compiled, not run on hardware
//
#include <hip/hip_runtime.h>
#include <math.h>

// ---------------------------------------------------------------- types
typedef __attribute__((ext_vector_type(16))) __bf16 v16bf;
typedef __attribute__((ext_vector_type(8)))  __bf16 v8bf;
typedef __attribute__((ext_vector_type(8)))  float  v8f;

#define TWO_PI_F 6.28318530717958647692f

__device__ __forceinline__ float gelu_f(float x) {
    return 0.5f * x * (1.0f + erff(x * 0.70710678118654752f));
}

__device__ __forceinline__ v8f v8f_zero() {
    v8f z;
#pragma unroll
    for (int r = 0; r < 8; ++r) z[r] = 0.0f;
    return z;
}

// ---- CDNA5 async global->LDS copy (VGPR-bypassing, tracked by ASYNCcnt)
__device__ __forceinline__ void async_copy_b128(unsigned lds_byte_addr, const void* gptr) {
    asm volatile("global_load_async_to_lds_b128 %0, %1, off"
                 :: "v"(lds_byte_addr), "v"(gptr)
                 : "memory");
}
__device__ __forceinline__ void wait_async_le4() {
    asm volatile("s_wait_asynccnt 4" ::: "memory");
}
__device__ __forceinline__ void wait_async_le0() {
    asm volatile("s_wait_asynccnt 0" ::: "memory");
}
// generic->LDS byte address: LDS aperture addresses truncate to the LDS offset
__device__ __forceinline__ unsigned lds_addr(const void* p) {
    return (unsigned)(size_t)p;
}

// A-fragment (16x32, 16-bit): lane = (l&15) picks M row, lane>>4 picks K half.
// halves 0..7  hold K = half*8 + j       (contiguous 8)
// halves 8..15 hold K = 16 + half*8 + j  (contiguous 8)
__device__ __forceinline__ v16bf load_frag_a(const __bf16* base, int ldk, int row, int hlf) {
    const v8bf* p0 = (const v8bf*)(base + (size_t)row * ldk + hlf * 8);
    const v8bf* p1 = (const v8bf*)(base + (size_t)row * ldk + 16 + hlf * 8);
    v8bf x0 = *p0, x1 = *p1;
    v16bf r;
#pragma unroll
    for (int j = 0; j < 8; ++j) { r[j] = x0[j]; r[8 + j] = x1[j]; }
    return r;
}

// B-fragment (32x16, 16-bit): lane = (l&15) picks N col, lane>>4 picks K half.
// LDS holds [N][K]; halves 0..7 hold K = half*8 + j, halves 8..15 hold K = 16 + half*8 + j.
__device__ __forceinline__ v16bf load_frag_b(const __bf16* base, int ldk, int col, int hlf) {
    const v8bf* p0 = (const v8bf*)(base + (size_t)col * ldk + hlf * 8);
    const v8bf* p1 = (const v8bf*)(base + (size_t)col * ldk + 16 + hlf * 8);
    v8bf x0 = *p0, x1 = *p1;
    v16bf r;
#pragma unroll
    for (int j = 0; j < 8; ++j) { r[j] = x0[j]; r[8 + j] = x1[j]; }
    return r;
}

// ---------------------------------------------------------------- WMMA GEMM
// C[M,N] = act(A[M,K](bf16) @ Wt[N,K](bf16,pre-transposed) + bias[N]) (+res)
// Double-buffered async LDS staging. ACT: 0 none, 1 GELU. RES: residual add.
// OUTBF: 1 -> write bf16, 0 -> write fp32.
template <int ACT, int RES, int OUTBF>
__global__ __launch_bounds__(256) void gemm_bf16_wmma(
    const __bf16* __restrict__ A, const __bf16* __restrict__ Wt,
    const float* __restrict__ bias, const float* __restrict__ res,
    void* __restrict__ Cout, int M, int N, int K) {
    constexpr int BM = 128, BN = 128, BK = 32;
    __shared__ __bf16 As[2][BM][BK];
    __shared__ __bf16 Bs[2][BN][BK];

    const int tid  = threadIdx.x;
    const int wv   = tid >> 5;
    const int lane = tid & 31;
    const int wm   = wv >> 2;            // 0..1 : 64-row slab
    const int wn   = wv & 3;             // 0..3 : 32-col slab
    const int m0   = blockIdx.y * BM;
    const int n0   = blockIdx.x * BN;
    const int hlf  = lane >> 4;
    const int l16  = lane & 15;
    const int r0   = tid >> 2;           // staging row (0..63), +64 for 2nd op
    const int sg   = (tid & 3) * 8;      // 16B segment within a 32-elem row

    v8f acc[4][2];
#pragma unroll
    for (int i = 0; i < 4; ++i)
#pragma unroll
        for (int j = 0; j < 2; ++j) acc[i][j] = v8f_zero();

    const __bf16* Abase = A + (size_t)m0 * K;
    const __bf16* Wbase = Wt + (size_t)n0 * K;

    // 4 async instructions per wave per tile (2 A-side, 2 B-side)
    auto issue = [&](int buf, int kt) {
        async_copy_b128(lds_addr(&As[buf][r0][sg]),       Abase + (size_t)r0 * K + kt + sg);
        async_copy_b128(lds_addr(&As[buf][64 + r0][sg]),  Abase + (size_t)(64 + r0) * K + kt + sg);
        async_copy_b128(lds_addr(&Bs[buf][r0][sg]),       Wbase + (size_t)r0 * K + kt + sg);
        async_copy_b128(lds_addr(&Bs[buf][64 + r0][sg]),  Wbase + (size_t)(64 + r0) * K + kt + sg);
    };

    const int nk = K / BK;
    issue(0, 0);
    int buf = 0;
    for (int it = 0; it < nk; ++it) {
        if (it + 1 < nk) {
            issue(buf ^ 1, (it + 1) * BK);
            wait_async_le4();          // previous tile's 4 ops retired
        } else {
            wait_async_le0();
        }
        __syncthreads();               // whole tile staged by all waves

        v16bf bf[2];
#pragma unroll
        for (int j = 0; j < 2; ++j)
            bf[j] = load_frag_b(&Bs[buf][0][0], BK, wn * 32 + j * 16 + l16, hlf);
#pragma unroll
        for (int i = 0; i < 4; ++i) {
            v16bf af = load_frag_a(&As[buf][0][0], BK, wm * 64 + i * 16 + l16, hlf);
#pragma unroll
            for (int j = 0; j < 2; ++j)
                acc[i][j] = __builtin_amdgcn_wmma_f32_16x16x32_bf16(
                    false, af, false, bf[j], (short)0, acc[i][j], false, false);
        }
        __syncthreads();               // done reading buf before it is refilled
        buf ^= 1;
    }

    // epilogue: C layout -> col = n0+..+l16 ; row = ..+hlf*8+r
#pragma unroll
    for (int i = 0; i < 4; ++i) {
        int rowb = m0 + wm * 64 + i * 16 + hlf * 8;
#pragma unroll
        for (int j = 0; j < 2; ++j) {
            int col = n0 + wn * 32 + j * 16 + l16;
            float bv = bias[col];
#pragma unroll
            for (int r = 0; r < 8; ++r) {
                float vv = acc[i][j][r] + bv;
                if (ACT == 1) vv = gelu_f(vv);
                size_t idx = (size_t)(rowb + r) * N + col;
                if (RES) vv += res[idx];
                if (OUTBF) ((__bf16*)Cout)[idx] = (__bf16)vv;
                else       ((float*)Cout)[idx]  = vv;
            }
        }
    }
}

// ---------------------------------------------------------------- LayerNorm (fp32 in, bf16 out)
template <int GELU_OUT>
__global__ __launch_bounds__(256) void layernorm_kernel(
    const float* __restrict__ x, const float* __restrict__ g,
    const float* __restrict__ b, __bf16* __restrict__ y, int D) {
    const int row  = blockIdx.x;
    const int tid  = threadIdx.x;
    const int wv   = tid >> 5;
    const int lane = tid & 31;
    const float* xr = x + (size_t)row * D;
    float s = 0.f, s2 = 0.f;
    for (int i = tid; i < D; i += 256) { float v = xr[i]; s += v; s2 += v * v; }
#pragma unroll
    for (int off = 16; off; off >>= 1) {
        s  += __shfl_down(s, off, 32);
        s2 += __shfl_down(s2, off, 32);
    }
    __shared__ float rs[8], rs2[8];
    if (lane == 0) { rs[wv] = s; rs2[wv] = s2; }
    __syncthreads();
    if (tid == 0) {
        float a = 0.f, a2 = 0.f;
        for (int w = 0; w < 8; ++w) { a += rs[w]; a2 += rs2[w]; }
        rs[0] = a; rs2[0] = a2;
    }
    __syncthreads();
    float mean = rs[0] / (float)D;
    float var  = rs2[0] / (float)D - mean * mean;
    float inv  = rsqrtf(var + 1e-5f);
    for (int i = tid; i < D; i += 256) {
        float v = (xr[i] - mean) * inv * g[i] + b[i];
        if (GELU_OUT) v = gelu_f(v);
        y[(size_t)row * D + i] = (__bf16)v;
    }
}

// ---------------------------------------------------------------- RoPE (in place, bf16 q & k)
__global__ void rope_kernel(__bf16* __restrict__ q, __bf16* __restrict__ k, int M) {
    int idx = blockIdx.x * blockDim.x + threadIdx.x;   // M * H * NF pairs
    if (idx >= M * 256) return;
    int row = idx >> 8;
    int p   = idx & 255;
    int h   = p >> 4;
    int jf  = p & 15;
    int pos = row & 1023;
    float inv = __powf(10000.0f, -((float)(2 * jf)) / 32.0f);
    float ang = (float)pos * inv;
    float c, s;
    __sincosf(ang, &s, &c);
    size_t base = (size_t)row * 1024 + h * 64 + 2 * jf;
    float x0 = (float)q[base], x1 = (float)q[base + 1];
    q[base]     = (__bf16)(x0 * c - x1 * s);
    q[base + 1] = (__bf16)(x1 * c + x0 * s);
    x0 = (float)k[base]; x1 = (float)k[base + 1];
    k[base]     = (__bf16)(x0 * c - x1 * s);
    k[base + 1] = (__bf16)(x1 * c + x0 * s);
}

// ---------------------------------------------------------------- flash attention (bf16 I/O)
// grid (L/QB, H, B), 256 threads. Each wave owns 32 query rows.
__global__ __launch_bounds__(256) void flash_attn(
    const __bf16* __restrict__ q, const __bf16* __restrict__ k,
    const __bf16* __restrict__ v, __bf16* __restrict__ o) {
    constexpr int Lk = 1024, Dh = 64, Dm = 1024, QB = 256, KT = 32;
    const float scale = 0.125f;   // 1/sqrt(64)

    __shared__ __align__(16) char smem[32768];
    __bf16* Qlds = (__bf16*)smem;                 // [256][64] (phase 1 only)
    __bf16* Kt   = (__bf16*)smem;                 // [32][64]  key-major
    __bf16* Vrow = (__bf16*)(smem + 4096);        // [32][64]  key-major (async landing)
    __bf16* Vt   = (__bf16*)(smem + 8192);        // [64][32]  dim-major
    __bf16* Pst  = (__bf16*)(smem + 12288);       // per-wave [32][32]

    const int tid  = threadIdx.x;
    const int wv   = tid >> 5;
    const int lane = tid & 31;
    const int hlf  = lane >> 4;
    const int l16  = lane & 15;
    const int b    = blockIdx.z, h = blockIdx.y;
    const int qb   = blockIdx.x * QB;

    const __bf16* qbase = q + (size_t)b * Lk * Dm + h * Dh;
    const __bf16* kbase = k + (size_t)b * Lk * Dm + h * Dh;
    const __bf16* vbase = v + (size_t)b * Lk * Dm + h * Dh;

    // ---- phase 1: async-stage Q block (thread t owns row t, 8x16B segments)
#pragma unroll
    for (int j = 0; j < 8; ++j)
        async_copy_b128(lds_addr(Qlds + tid * Dh + j * 8),
                        qbase + (size_t)(qb + tid) * Dm + j * 8);
    wait_async_le0();
    __syncthreads();
    v16bf qf[2][2];
#pragma unroll
    for (int mi = 0; mi < 2; ++mi)
#pragma unroll
        for (int c = 0; c < 2; ++c)
            qf[mi][c] = load_frag_a(Qlds + 32 * c, Dh, wv * 32 + mi * 16 + l16, hlf);
    __syncthreads();   // Qlds space is reused below

    v8f oacc[2][4];
    float mrow[2][8], lrow[2][8];
#pragma unroll
    for (int mi = 0; mi < 2; ++mi) {
#pragma unroll
        for (int nd = 0; nd < 4; ++nd) oacc[mi][nd] = v8f_zero();
#pragma unroll
        for (int r = 0; r < 8; ++r) { mrow[mi][r] = -1e30f; lrow[mi][r] = 0.f; }
    }

    const int sr = tid >> 3;          // staging row 0..31
    const int sc = (tid & 7) * 8;     // 16B segment within 64-elem row

    for (int kt0 = 0; kt0 < Lk; kt0 += KT) {
        // async-stage K and V tiles (raw bf16, one b128 per thread per tensor)
        async_copy_b128(lds_addr(Kt + sr * Dh + sc),   kbase + (size_t)(kt0 + sr) * Dm + sc);
        async_copy_b128(lds_addr(Vrow + sr * Dh + sc), vbase + (size_t)(kt0 + sr) * Dm + sc);
        wait_async_le0();
        __syncthreads();
        // transpose V tile in LDS: Vt[d][key] = Vrow[key][d]
        for (int e = tid; e < KT * Dh; e += 256) {
            int key = e >> 6, d = e & 63;
            Vt[d * KT + key] = Vrow[key * Dh + d];
        }
        __syncthreads();

        // S = scale * Q K^T   (two d-chunks of 32 chained per tile)
        v8f s[2][2];
#pragma unroll
        for (int mi = 0; mi < 2; ++mi)
#pragma unroll
            for (int nt = 0; nt < 2; ++nt) {
                v16bf kf0 = load_frag_b(Kt + 0,  Dh, nt * 16 + l16, hlf);
                v16bf kf1 = load_frag_b(Kt + 32, Dh, nt * 16 + l16, hlf);
                v8f t = __builtin_amdgcn_wmma_f32_16x16x32_bf16(
                    false, qf[mi][0], false, kf0, (short)0, v8f_zero(), false, false);
                t = __builtin_amdgcn_wmma_f32_16x16x32_bf16(
                    false, qf[mi][1], false, kf1, (short)0, t, false, false);
#pragma unroll
                for (int r = 0; r < 8; ++r) t[r] *= scale;
                s[mi][nt] = t;
            }

        // online softmax (row stats via 16-lane xor reductions)
#pragma unroll
        for (int mi = 0; mi < 2; ++mi) {
            float tm[8], nm[8], al[8], ps[8];
#pragma unroll
            for (int r = 0; r < 8; ++r) tm[r] = fmaxf(s[mi][0][r], s[mi][1][r]);
#pragma unroll
            for (int off = 8; off; off >>= 1)
#pragma unroll
                for (int r = 0; r < 8; ++r)
                    tm[r] = fmaxf(tm[r], __shfl_xor(tm[r], off, 32));
#pragma unroll
            for (int r = 0; r < 8; ++r) {
                nm[r] = fmaxf(mrow[mi][r], tm[r]);
                al[r] = __expf(mrow[mi][r] - nm[r]);
                mrow[mi][r] = nm[r];
                ps[r] = 0.f;
            }
#pragma unroll
            for (int nt = 0; nt < 2; ++nt)
#pragma unroll
                for (int r = 0; r < 8; ++r) {
                    float pe = __expf(s[mi][nt][r] - nm[r]);
                    ps[r] += pe;
                    Pst[(size_t)wv * 1024 + (mi * 16 + hlf * 8 + r) * 32 + nt * 16 + l16] =
                        (__bf16)pe;
                }
#pragma unroll
            for (int off = 8; off; off >>= 1)
#pragma unroll
                for (int r = 0; r < 8; ++r) ps[r] += __shfl_xor(ps[r], off, 32);
#pragma unroll
            for (int r = 0; r < 8; ++r) lrow[mi][r] = lrow[mi][r] * al[r] + ps[r];
#pragma unroll
            for (int nd = 0; nd < 4; ++nd)
#pragma unroll
                for (int r = 0; r < 8; ++r) oacc[mi][nd][r] *= al[r];
        }

        // O += P @ V   (K = 32 keys)
#pragma unroll
        for (int mi = 0; mi < 2; ++mi) {
            v16bf pf = load_frag_a(Pst + (size_t)wv * 1024, 32, mi * 16 + l16, hlf);
#pragma unroll
            for (int nd = 0; nd < 4; ++nd) {
                v16bf vf = load_frag_b(Vt, KT, nd * 16 + l16, hlf);
                oacc[mi][nd] = __builtin_amdgcn_wmma_f32_16x16x32_bf16(
                    false, pf, false, vf, (short)0, oacc[mi][nd], false, false);
            }
        }
        __syncthreads();   // done with Kt/Vrow/Vt before next tile overwrite
    }

    // normalize + store (C layout), bf16 output feeds the O-projection GEMM
#pragma unroll
    for (int mi = 0; mi < 2; ++mi) {
        float inv[8];
#pragma unroll
        for (int r = 0; r < 8; ++r) inv[r] = 1.0f / lrow[mi][r];
#pragma unroll
        for (int nd = 0; nd < 4; ++nd)
#pragma unroll
            for (int r = 0; r < 8; ++r) {
                int grow = qb + wv * 32 + mi * 16 + hlf * 8 + r;
                int gcol = h * Dh + nd * 16 + l16;
                o[((size_t)b * Lk + grow) * Dm + gcol] = (__bf16)(oacc[mi][nd][r] * inv[r]);
            }
    }
}

// ---------------------------------------------------------------- conversion / small kernels
// w: count x [K][N] fp32 -> wt: count x [N][K] bf16 (transpose for B-frag K-contiguity)
__global__ void transpose_bf16_kernel(const float* __restrict__ w, __bf16* __restrict__ wt,
                                      int K, int N, int count) {
    size_t i = (size_t)blockIdx.x * 256 + threadIdx.x;
    size_t kn = (size_t)K * N;
    if (i >= kn * count) return;
    size_t l = i / kn, rem = i % kn;
    int kk = (int)(rem / N), n = (int)(rem % N);
    wt[l * kn + (size_t)n * K + kk] = (__bf16)w[i];
}

__global__ void f32_to_bf16_kernel(const float* __restrict__ s, __bf16* __restrict__ d, int n) {
    int i = blockIdx.x * 256 + threadIdx.x;
    if (i < n) d[i] = (__bf16)s[i];
}

__global__ void precond_kernel(const float* __restrict__ x_t, const float* __restrict__ sigmas,
                               const float* __restrict__ sin_w,
                               __bf16* __restrict__ xin, __bf16* __restrict__ tfeat) {
    int m = blockIdx.x;
    int j = threadIdx.x;   // 0..159
    float sg  = sigmas[m];
    float cin = rsqrtf(sg * sg + 1.0f);
    float t   = 0.25f * logf(sg + 0.001f);
    if (j < 64) xin[(size_t)m * 64 + j] = (__bf16)(cin * x_t[(size_t)m * 64 + j]);
    float val;
    if (j == 0)        val = t;
    else if (j <= 64)  val = sinf(t * sin_w[j - 1] * TWO_PI_F);
    else if (j <= 128) val = cosf(t * sin_w[j - 65] * TWO_PI_F);
    else               val = 0.0f;
    tfeat[(size_t)m * 160 + j] = (__bf16)val;
}

// tm1_w [129][128] -> bf16 [N=128][K=160] zero-padded
__global__ void pad_tm1_kernel(const float* __restrict__ w, __bf16* __restrict__ wt) {
    int i = blockIdx.x * 256 + threadIdx.x;      // 128*160
    if (i >= 128 * 160) return;
    int n = i / 160, kk = i % 160;
    wt[i] = (__bf16)((kk < 129) ? w[kk * 128 + n] : 0.0f);
}

// o2_w [1024][64] -> bf16 [N=128][K=1024] zero-padded; bias 64 -> 128 fp32
__global__ void pad_o2_kernel(const float* __restrict__ w, const float* __restrict__ bs,
                              __bf16* __restrict__ wt, float* __restrict__ bo) {
    int i = blockIdx.x * 256 + threadIdx.x;      // 128*1024
    if (i < 128) bo[i] = (i < 64) ? bs[i] : 0.0f;
    if (i >= 128 * 1024) return;
    int n = i >> 10, kk = i & 1023;
    wt[i] = (__bf16)((n < 64) ? w[kk * 64 + n] : 0.0f);
}

__global__ void final_kernel(const float* __restrict__ mo, const float* __restrict__ x_t,
                             const float* __restrict__ sigmas, float* __restrict__ out) {
    int i = blockIdx.x * 256 + threadIdx.x;      // 4096*64
    if (i >= 4096 * 64) return;
    int m = i >> 6, e = i & 63;
    float s     = sigmas[m];
    float cskip = 1.0f / (s * s + 1.0f);
    float cout  = s * rsqrtf(s * s + 1.0f);
    out[i] = cout * mo[(size_t)m * 128 + e] + cskip * x_t[i];
}

// ---------------------------------------------------------------- launch
extern "C" void kernel_launch(void* const* d_in, const int* in_sizes, int n_in,
                              void* d_out, int out_size, void* d_ws, size_t ws_size,
                              hipStream_t stream) {
    constexpr int M = 4096, D = 1024, E = 64, NL = 8;

    const float* x_t    = (const float*)d_in[0];
    const float* sigmas = (const float*)d_in[1];
    const float* sin_w  = (const float*)d_in[2];
    const float* tm1_w  = (const float*)d_in[3];
    const float* tm1_b  = (const float*)d_in[4];
    const float* tm2_w  = (const float*)d_in[5];
    const float* tm2_b  = (const float*)d_in[6];
    const float* pr1_w  = (const float*)d_in[7];
    const float* pr1_b  = (const float*)d_in[8];
    const float* pr2_w  = (const float*)d_in[9];
    const float* pr2_b  = (const float*)d_in[10];
    const float* ln1_g  = (const float*)d_in[11];
    const float* ln1_b  = (const float*)d_in[12];
    const float* wq_w   = (const float*)d_in[13];
    const float* wq_b   = (const float*)d_in[14];
    const float* wk_w   = (const float*)d_in[15];
    const float* wk_b   = (const float*)d_in[16];
    const float* wv_w   = (const float*)d_in[17];
    const float* wv_b   = (const float*)d_in[18];
    const float* wo_w   = (const float*)d_in[19];
    const float* wo_b   = (const float*)d_in[20];
    const float* ln2_g  = (const float*)d_in[21];
    const float* ln2_b  = (const float*)d_in[22];
    const float* f1_w   = (const float*)d_in[23];
    const float* f1_b   = (const float*)d_in[24];
    const float* f2_w   = (const float*)d_in[25];
    const float* f2_b   = (const float*)d_in[26];
    const float* o1_w   = (const float*)d_in[27];
    const float* o1_b   = (const float*)d_in[28];
    const float* oln_g  = (const float*)d_in[29];
    const float* oln_b  = (const float*)d_in[30];
    const float* o2_w   = (const float*)d_in[31];
    const float* o2_b   = (const float*)d_in[32];

    // deterministic workspace carve
    char* p = (char*)d_ws;
    auto alloc = [&](size_t bytes) -> void* {
        void* r = (void*)p;
        p += (bytes + 255) & ~(size_t)255;
        return r;
    };
    // bf16 pre-transposed weights [N][K]
    __bf16* tm1t = (__bf16*)alloc((size_t)128 * 160 * 2);
    __bf16* tm2t = (__bf16*)alloc((size_t)D * 128 * 2);
    __bf16* pr1t = (__bf16*)alloc((size_t)D * E * 2);
    __bf16* pr2t = (__bf16*)alloc((size_t)D * D * 2);
    __bf16* wqt  = (__bf16*)alloc((size_t)NL * D * D * 2);
    __bf16* wkt  = (__bf16*)alloc((size_t)NL * D * D * 2);
    __bf16* wvt  = (__bf16*)alloc((size_t)NL * D * D * 2);
    __bf16* wot  = (__bf16*)alloc((size_t)NL * D * D * 2);
    __bf16* f1t  = (__bf16*)alloc((size_t)NL * D * 4 * D * 2);
    __bf16* f2t  = (__bf16*)alloc((size_t)NL * 4 * D * D * 2);
    __bf16* o1t  = (__bf16*)alloc((size_t)D * D * 2);
    __bf16* o2t  = (__bf16*)alloc((size_t)128 * D * 2);
    float*  o2bp = (float*)alloc((size_t)128 * 4);
    // activations
    __bf16* tfeat = (__bf16*)alloc((size_t)M * 160 * 2);
    __bf16* tembh = (__bf16*)alloc((size_t)M * 128 * 2);
    float*  temb  = (float*)alloc((size_t)M * D * 4);
    __bf16* xin   = (__bf16*)alloc((size_t)M * E * 2);
    float*  hbuf  = (float*)alloc((size_t)M * D * 4);
    __bf16* ybuf  = (__bf16*)alloc((size_t)M * D * 2);
    __bf16* qb    = (__bf16*)alloc((size_t)M * D * 2);
    __bf16* kb    = (__bf16*)alloc((size_t)M * D * 2);
    __bf16* vb    = (__bf16*)alloc((size_t)M * D * 2);
    __bf16* aob   = (__bf16*)alloc((size_t)M * D * 2);
    __bf16* mlph  = (__bf16*)alloc((size_t)M * 4 * D * 2);
    float*  hob   = (float*)alloc((size_t)M * D * 4);
    __bf16* hbf   = (__bf16*)alloc((size_t)M * D * 2);
    float*  mob   = (float*)alloc((size_t)M * 128 * 4);

    dim3 blk256(256);
    auto gemm_grid = [](int N) { return dim3(N / 128, M / 128); };
    auto cgrid = [](size_t n) { return dim3((unsigned)((n + 255) / 256)); };

    // ---- one-time-per-launch weight conversion (fp32 -> bf16, transposed)
    transpose_bf16_kernel<<<cgrid((size_t)128 * D), blk256, 0, stream>>>(tm2_w, tm2t, 128, D, 1);
    transpose_bf16_kernel<<<cgrid((size_t)E * D), blk256, 0, stream>>>(pr1_w, pr1t, E, D, 1);
    transpose_bf16_kernel<<<cgrid((size_t)D * D), blk256, 0, stream>>>(pr2_w, pr2t, D, D, 1);
    transpose_bf16_kernel<<<cgrid((size_t)NL * D * D), blk256, 0, stream>>>(wq_w, wqt, D, D, NL);
    transpose_bf16_kernel<<<cgrid((size_t)NL * D * D), blk256, 0, stream>>>(wk_w, wkt, D, D, NL);
    transpose_bf16_kernel<<<cgrid((size_t)NL * D * D), blk256, 0, stream>>>(wv_w, wvt, D, D, NL);
    transpose_bf16_kernel<<<cgrid((size_t)NL * D * D), blk256, 0, stream>>>(wo_w, wot, D, D, NL);
    transpose_bf16_kernel<<<cgrid((size_t)NL * D * 4 * D), blk256, 0, stream>>>(f1_w, f1t, D, 4 * D, NL);
    transpose_bf16_kernel<<<cgrid((size_t)NL * 4 * D * D), blk256, 0, stream>>>(f2_w, f2t, 4 * D, D, NL);
    transpose_bf16_kernel<<<cgrid((size_t)D * D), blk256, 0, stream>>>(o1_w, o1t, D, D, 1);
    pad_tm1_kernel<<<cgrid(128 * 160), blk256, 0, stream>>>(tm1_w, tm1t);
    pad_o2_kernel<<<cgrid(128 * 1024), blk256, 0, stream>>>(o2_w, o2_b, o2t, o2bp);

    // ---- preconditioning
    precond_kernel<<<M, 160, 0, stream>>>(x_t, sigmas, sin_w, xin, tfeat);

    // ---- time embedding MLP
    gemm_bf16_wmma<1, 0, 1><<<gemm_grid(128), blk256, 0, stream>>>(
        tfeat, tm1t, tm1_b, nullptr, tembh, M, 128, 160);
    gemm_bf16_wmma<0, 0, 0><<<gemm_grid(D), blk256, 0, stream>>>(
        tembh, tm2t, tm2_b, nullptr, temb, M, D, 128);

    // ---- input projection + temb
    gemm_bf16_wmma<1, 0, 1><<<gemm_grid(D), blk256, 0, stream>>>(
        xin, pr1t, pr1_b, nullptr, ybuf, M, D, E);
    gemm_bf16_wmma<0, 1, 0><<<gemm_grid(D), blk256, 0, stream>>>(
        ybuf, pr2t, pr2_b, temb, hbuf, M, D, D);

    // ---- transformer layers
    for (int i = 0; i < NL; ++i) {
        const size_t wD2 = (size_t)i * D * D;
        const size_t wD  = (size_t)i * D;
        const size_t wF2 = (size_t)i * D * 4 * D;
        const size_t wF  = (size_t)i * 4 * D;

        layernorm_kernel<0><<<M, blk256, 0, stream>>>(hbuf, ln1_g + wD, ln1_b + wD, ybuf, D);

        gemm_bf16_wmma<0, 0, 1><<<gemm_grid(D), blk256, 0, stream>>>(
            ybuf, wqt + wD2, wq_b + wD, nullptr, qb, M, D, D);
        gemm_bf16_wmma<0, 0, 1><<<gemm_grid(D), blk256, 0, stream>>>(
            ybuf, wkt + wD2, wk_b + wD, nullptr, kb, M, D, D);
        gemm_bf16_wmma<0, 0, 1><<<gemm_grid(D), blk256, 0, stream>>>(
            ybuf, wvt + wD2, wv_b + wD, nullptr, vb, M, D, D);

        rope_kernel<<<cgrid((size_t)M * 256), blk256, 0, stream>>>(qb, kb, M);

        flash_attn<<<dim3(4, 16, 4), blk256, 0, stream>>>(qb, kb, vb, aob);

        gemm_bf16_wmma<0, 1, 0><<<gemm_grid(D), blk256, 0, stream>>>(
            aob, wot + wD2, wo_b + wD, hbuf, hbuf, M, D, D);

        layernorm_kernel<0><<<M, blk256, 0, stream>>>(hbuf, ln2_g + wD, ln2_b + wD, ybuf, D);

        gemm_bf16_wmma<1, 0, 1><<<gemm_grid(4 * D), blk256, 0, stream>>>(
            ybuf, f1t + wF2, f1_b + wF, nullptr, mlph, M, 4 * D, D);
        gemm_bf16_wmma<0, 1, 0><<<gemm_grid(D), blk256, 0, stream>>>(
            mlph, f2t + wF2, f2_b + wD, hbuf, hbuf, M, D, 4 * D);
    }

    // ---- output head
    f32_to_bf16_kernel<<<cgrid((size_t)M * D), blk256, 0, stream>>>(hbuf, hbf, M * D);
    gemm_bf16_wmma<0, 0, 0><<<gemm_grid(D), blk256, 0, stream>>>(
        hbf, o1t, o1_b, nullptr, hob, M, D, D);
    layernorm_kernel<1><<<M, blk256, 0, stream>>>(hob, oln_g, oln_b, ybuf, D);
    gemm_bf16_wmma<0, 0, 0><<<gemm_grid(128), blk256, 0, stream>>>(
        ybuf, o2t, o2bp, nullptr, mob, M, 128, D);

    final_kernel<<<cgrid((size_t)M * E), blk256, 0, stream>>>(mob, x_t, sigmas, (float*)d_out);
}
